// PairformerLayer_21440476741695
// MI455X (gfx1250) — matrix operand order repaired and checked by reading, not compiled, not run on hardware
//
#include <hip/hip_runtime.h>

// ---------------------------------------------------------------------------
// Pairformer layer for MI455X (gfx1250, wave32). All GEMM-shaped math runs on
// v_wmma_f32_16x16x32_bf16 (bf16 inputs, f32 accumulate). Elementwise / LN /
// softmax bookkeeping stays in f32 VALU.
// ---------------------------------------------------------------------------

typedef __attribute__((ext_vector_type(16))) __bf16 v16bf;
typedef __attribute__((ext_vector_type(8)))  __bf16 v8bf;
typedef __attribute__((ext_vector_type(8)))  float  v8f;

static __device__ __forceinline__ v8f wmma_bf16(v16bf a, v16bf b, v8f c) {
    return __builtin_amdgcn_wmma_f32_16x16x32_bf16(false, a, false, b,
                                                   (short)0, c, false, false);
}

static __device__ __forceinline__ float sigmf(float x) {
    return 1.f / (1.f + __expf(-x));
}

// Epilogue modes for the generic GEMM
#define EP_F32        0   // c0[m*ldc+n] = acc
#define EP_BF16       1   // c1[m*ldc+n] = bf16(acc + bias[n])
#define EP_SIG_BF16   2   // c1 = bf16(sigmoid(acc))
#define EP_SILU_BF16  3   // c1 = bf16(acc*sigmoid(acc))
#define EP_MUL_BF16   4   // c1 = bf16(acc * e1[m*lde+n])
#define EP_GATE_CM    5   // channel-major gated split-store (tri-mul a/b)
#define EP_RESID      6   // c0[m*ldc+n] += acc
#define EP_RESID_T    7   // c0[swap(m)*ldc+n] += acc   (ending tri-attn)
#define EP_GATE_RESID 8   // c0 += acc * sigmoid(e0[m*lde+n])

static __device__ __forceinline__ void
gemm_epilogue(int ep, const v8f& acc, int mrbase, int hv, int n, int Nc,
              size_t cb, float* c0, __bf16* c1, __bf16* c2,
              const float* e0, const __bf16* e1, const float* biasv,
              int ldc, int lde, int tflag)
{
    if (n >= Nc) return;
    #pragma unroll
    for (int vv = 0; vv < 8; vv++) {
        int mr = mrbase + vv + (hv ? 8 : 0);     // C row (ISA C-layout)
        float x = acc[vv];
        switch (ep) {
        case EP_F32:  c0[cb + (size_t)mr * ldc + n] = x; break;
        case EP_BF16: {
            float bv = biasv ? biasv[n] : 0.f;
            c1[(size_t)mr * ldc + n] = (__bf16)(x + bv);
        } break;
        case EP_SIG_BF16:  c1[(size_t)mr * ldc + n] = (__bf16)sigmf(x); break;
        case EP_SILU_BF16: c1[(size_t)mr * ldc + n] = (__bf16)(x * sigmf(x)); break;
        case EP_MUL_BF16:
            c1[(size_t)mr * ldc + n] =
                (__bf16)(x * (float)e1[(size_t)mr * lde + n]);
            break;
        case EP_GATE_CM: {  // gated projection -> channel-major a/b operands
            float g   = (float)e1[(size_t)mr * lde + n];  // pre-sigmoided gate
            int   row = mr >> 8, col = mr & 255;          // m = row*256+col
            int   c   = n & 127;
            __bf16* dst = (n < 128) ? c1 : c2;
            size_t di = (size_t)c * 65536 +
                        (tflag ? (size_t)col * 256 + row
                               : (size_t)row * 256 + col);
            dst[di] = (__bf16)(x * g);
        } break;
        case EP_RESID: c0[(size_t)mr * ldc + n] += x; break;
        case EP_RESID_T: {
            size_t mt = (size_t)(mr & 255) * 256 + (mr >> 8);
            c0[mt * ldc + n] += x;
        } break;
        case EP_GATE_RESID: {
            float g = e0[(size_t)mr * lde + n];
            c0[(size_t)mr * ldc + n] += x * sigmf(g);
        } break;
        }
    }
}

// ---------------------------------------------------------------------------
// Generic (optionally batched) bf16 WMMA GEMM, 2x2 register blocking:
// one wave computes a 32x32 tile (4 WMMAs per K-step, each A/B fragment
// reused twice). M must be a multiple of 32 (true for all call sites).
//   A  : [M][K] row-major bf16, leading dim lda
//   Bt : [Nc][K] (i.e. B^T) row-major bf16, leading dim K
// Out-of-range B columns are clamped (loaded garbage only feeds lanes whose
// results are discarded in the epilogue) -> no EXEC divergence in the K-loop.
// A/B k order is contiguous (lane half h: k 16h..16h+15); C uses the ISA layout.
// ---------------------------------------------------------------------------
__global__ void gemm_bf16_wmma(const __bf16* __restrict__ A,
                               const __bf16* __restrict__ Bt,
                               int M, int Nc, int K, int lda,
                               long long sA, long long sB, long long sC,
                               float* c0, __bf16* c1, __bf16* c2,
                               const float* e0, const __bf16* e1,
                               const float* biasv,
                               int ldc, int lde, int ep, int tflag)
{
    int nTn  = (Nc + 31) >> 5;
    int nT   = (M >> 5) * nTn;
    int wid  = threadIdx.x >> 5;
    int tile = blockIdx.x * 8 + wid;
    if (tile >= nT) return;                  // wave-uniform exit
    int tm = tile / nTn, tn = tile % nTn;
    int lane = threadIdx.x & 31;
    int hv   = lane >> 4;                    // lane half (0/1)
    int l16  = lane & 15;

    const __bf16* Ab = A  + (size_t)blockIdx.y * (size_t)sA;
    const __bf16* Bb = Bt + (size_t)blockIdx.y * (size_t)sB;

    int m0 = tm * 32 + l16;                  // A rows for this lane
    int m1 = m0 + 16;
    int n0 = tn * 32 + l16;                  // B cols for this lane
    int n1 = n0 + 16;
    int nc0 = (n0 < Nc) ? n0 : (Nc - 1);     // clamped (garbage discarded)
    int nc1 = (n1 < Nc) ? n1 : (Nc - 1);

    const __bf16* ap0 = Ab + (size_t)m0 * lda + (hv ? 16 : 0);
    const __bf16* ap1 = Ab + (size_t)m1 * lda + (hv ? 16 : 0);
    const __bf16* bp0 = Bb + (size_t)nc0 * K + (hv ? 16 : 0);
    const __bf16* bp1 = Bb + (size_t)nc1 * K + (hv ? 16 : 0);

    v8f acc00 = {}, acc01 = {}, acc10 = {}, acc11 = {};
    for (int k0 = 0; k0 < K; k0 += 32) {
        v16bf a0, a1;
        {   // A frags: lane half 0 -> k0..k0+15 ; half 1 -> +16
            v8bf x1 = *(const v8bf*)(ap0 + k0);
            v8bf x2 = *(const v8bf*)(ap0 + k0 + 8);
            v8bf y1 = *(const v8bf*)(ap1 + k0);
            v8bf y2 = *(const v8bf*)(ap1 + k0 + 8);
            #pragma unroll
            for (int i = 0; i < 8; i++) {
                a0[i] = x1[i]; a0[8 + i] = x2[i];
                a1[i] = y1[i]; a1[8 + i] = y2[i];
            }
        }
        // B frags: lanes 0-15 hold k0..k0+15, lanes 16-31 +16
        v16bf b0 = *(const v16bf*)(bp0 + k0);
        v16bf b1 = *(const v16bf*)(bp1 + k0);
        acc00 = wmma_bf16(a0, b0, acc00);
        acc01 = wmma_bf16(a0, b1, acc01);
        acc10 = wmma_bf16(a1, b0, acc10);
        acc11 = wmma_bf16(a1, b1, acc11);
    }

    size_t cb = (size_t)blockIdx.y * (size_t)sC;
    gemm_epilogue(ep, acc00, tm * 32,      hv, n0, Nc, cb, c0, c1, c2,
                  e0, e1, biasv, ldc, lde, tflag);
    gemm_epilogue(ep, acc01, tm * 32,      hv, n1, Nc, cb, c0, c1, c2,
                  e0, e1, biasv, ldc, lde, tflag);
    gemm_epilogue(ep, acc10, tm * 32 + 16, hv, n0, Nc, cb, c0, c1, c2,
                  e0, e1, biasv, ldc, lde, tflag);
    gemm_epilogue(ep, acc11, tm * 32 + 16, hv, n1, Nc, cb, c0, c1, c2,
                  e0, e1, biasv, ldc, lde, tflag);
}

// ---------------------------------------------------------------------------
// Fused bias-attention kernel (softmax over 256 keys), WMMA for QK^T and PV.
// One workgroup = one (outer,head). D padded to 32 (Dreal may be 24).
// Two-pass stable softmax; P tiles staged through LDS to re-layout into the
// WMMA A-operand format.
// ---------------------------------------------------------------------------
__global__ void attn_kernel(const __bf16* __restrict__ q,
                            const __bf16* __restrict__ k,
                            const __bf16* __restrict__ v,
                            int ld, long long outerStride, long long headStride,
                            int H, int Dreal, float scale,
                            const float* __restrict__ bias, int biasLd,
                            float* __restrict__ out,
                            long long oOS, long long oHS, int outLd)
{
    constexpr int Nk = 256;
    __shared__ __attribute__((aligned(64))) __bf16 Qs[Nk * 32];
    __shared__ __attribute__((aligned(64))) __bf16 Ks[Nk * 32];
    __shared__ __attribute__((aligned(64))) __bf16 Vt[32 * Nk];   // [d][key]
    __shared__ __attribute__((aligned(64))) __bf16 Ps[8][16 * 32];

    int h = blockIdx.x % H;
    long long outer = blockIdx.x / H;
    size_t base = (size_t)outer * outerStride + (size_t)h * headStride;
    int tid = threadIdx.x;

    for (int idx = tid; idx < Nk * 32; idx += 256) {
        int r = idx >> 5, d = idx & 31;
        __bf16 qv = (d < Dreal) ? q[base + (size_t)r * ld + d] : (__bf16)0.f;
        __bf16 kv = (d < Dreal) ? k[base + (size_t)r * ld + d] : (__bf16)0.f;
        __bf16 vv = (d < Dreal) ? v[base + (size_t)r * ld + d] : (__bf16)0.f;
        Qs[idx] = qv; Ks[idx] = kv; Vt[d * Nk + r] = vv;
    }
    __syncthreads();

    int wid = tid >> 5, lane = tid & 31;
    int hv = lane >> 4, l16 = lane & 15;

    for (int jt = wid; jt < 16; jt += 8) {
        // Q fragment for this row tile (D = 32 -> single K-step)
        v16bf aq;
        {
            int j  = jt * 16 + l16;
            int d0 = hv ? 16 : 0;
            v8bf c1 = *(const v8bf*)&Qs[j * 32 + d0];
            v8bf c2 = *(const v8bf*)&Qs[j * 32 + d0 + 8];
            #pragma unroll
            for (int i = 0; i < 8; i++) { aq[i] = c1[i]; aq[8 + i] = c2[i]; }
        }
        // ---- pass 1: row max ----
        float mx[8];
        #pragma unroll
        for (int vv = 0; vv < 8; vv++) mx[vv] = -1e30f;
        for (int kt = 0; kt < 16; kt++) {
            int kk = kt * 16 + l16;
            v16bf bk = *(const v16bf*)&Ks[kk * 32 + (hv ? 16 : 0)];
            v8f acc = {};
            acc = wmma_bf16(aq, bk, acc);
            #pragma unroll
            for (int vv = 0; vv < 8; vv++) {
                int jr = jt * 16 + vv + (hv ? 8 : 0);
                int kc = kt * 16 + l16;
                float sc = acc[vv] * scale +
                           bias[((size_t)jr * Nk + kc) * biasLd + h];
                for (int o = 1; o < 16; o <<= 1)
                    sc = fmaxf(sc, __shfl_xor(sc, o, 32));
                mx[vv] = fmaxf(mx[vv], sc);
            }
        }
        // ---- pass 2: exp, row sums, P*V ----
        float ls[8];
        #pragma unroll
        for (int vv = 0; vv < 8; vv++) ls[vv] = 0.f;
        v8f o0 = {}, o1 = {};
        for (int kp = 0; kp < 8; kp++) {
            #pragma unroll
            for (int hh = 0; hh < 2; hh++) {
                int kt = kp * 2 + hh;
                int kk = kt * 16 + l16;
                v16bf bk = *(const v16bf*)&Ks[kk * 32 + (hv ? 16 : 0)];
                v8f acc = {};
                acc = wmma_bf16(aq, bk, acc);
                #pragma unroll
                for (int vv = 0; vv < 8; vv++) {
                    int jr = jt * 16 + vv + (hv ? 8 : 0);
                    int kc = kt * 16 + l16;
                    float sc = acc[vv] * scale +
                               bias[((size_t)jr * Nk + kc) * biasLd + h];
                    float pr = __expf(sc - mx[vv]);
                    float ssum = pr;
                    for (int o = 1; o < 16; o <<= 1)
                        ssum += __shfl_xor(ssum, o, 32);
                    ls[vv] += ssum;
                    Ps[wid][(vv + (hv ? 8 : 0)) * 32 + hh * 16 + l16] =
                        (__bf16)pr;
                }
            }
            // reload P tile in WMMA A-operand layout
            v16bf ap;
            {
                int r   = l16;
                int lk0 = hv ? 16 : 0;
                v8bf c1 = *(const v8bf*)&Ps[wid][r * 32 + lk0];
                v8bf c2 = *(const v8bf*)&Ps[wid][r * 32 + lk0 + 8];
                #pragma unroll
                for (int i = 0; i < 8; i++) { ap[i] = c1[i]; ap[8+i] = c2[i]; }
            }
            int kbase = kp * 32;
            v16bf bv0 = *(const v16bf*)&Vt[(l16)      * Nk + kbase + (hv ? 16 : 0)];
            v16bf bv1 = *(const v16bf*)&Vt[(16 + l16) * Nk + kbase + (hv ? 16 : 0)];
            o0 = wmma_bf16(ap, bv0, o0);
            o1 = wmma_bf16(ap, bv1, o1);
        }
        // ---- write normalized output ----
        #pragma unroll
        for (int vv = 0; vv < 8; vv++) {
            int jr = jt * 16 + vv + (hv ? 8 : 0);
            float inv = 1.f / ls[vv];
            size_t ob = (size_t)outer * oOS + (size_t)h * oHS +
                        (size_t)jr * outLd;
            int d0 = l16;
            if (d0 < Dreal) out[ob + d0] = o0[vv] * inv;
            int d1 = 16 + l16;
            if (d1 < Dreal) out[ob + d1] = o1[vv] * inv;
        }
    }
}

// ---------------------------------------------------------------------------
// LayerNorm: one wave per row, bf16 output. mode 0: in[m*C+c];
// mode 1: transposed pair read (m over 256x256); mode 2: channel-major
// in[c*65536+m] (triangle-mul intermediate).
// ---------------------------------------------------------------------------
__global__ void ln_kernel(const float* __restrict__ in,
                          const float* __restrict__ w,
                          const float* __restrict__ b,
                          __bf16* __restrict__ out,
                          int M, int C, int mode)
{
    int row = blockIdx.x * 8 + (threadIdx.x >> 5);
    if (row >= M) return;
    int lane = threadIdx.x & 31;
    int nIt = C >> 5;          // 4 (C=128) or 12 (C=384)
    float buf[12];
    size_t base, stride;
    if (mode == 0)      { base = (size_t)row * C; stride = 1; }
    else if (mode == 1) { size_t rt = (size_t)(row & 255) * 256 + (row >> 8);
                          base = rt * C; stride = 1; }
    else                { base = (size_t)row; stride = 65536; }
    float sum = 0.f;
    for (int i = 0; i < nIt; i++) {
        float x = in[base + (size_t)(lane + i * 32) * stride];
        buf[i] = x; sum += x;
    }
    for (int o = 1; o < 32; o <<= 1) sum += __shfl_xor(sum, o, 32);
    float mu = sum / C;
    float var = 0.f;
    for (int i = 0; i < nIt; i++) { float d = buf[i] - mu; var += d * d; }
    for (int o = 1; o < 32; o <<= 1) var += __shfl_xor(var, o, 32);
    float rinv = rsqrtf(var / C + 1e-5f);
    for (int i = 0; i < nIt; i++) {
        int c = lane + i * 32;
        out[(size_t)row * C + c] = (__bf16)((buf[i] - mu) * rinv * w[c] + b[c]);
    }
}

// f32 weight [K][Nc] -> bf16 transposed [Nc][K]
__global__ void wt_kernel(const float* __restrict__ w, __bf16* __restrict__ out,
                          int K, int Nc)
{
    int i = blockIdx.x * 256 + threadIdx.x;
    if (i < K * Nc) {
        int n = i / K, kk = i % K;
        out[i] = (__bf16)w[(size_t)kk * Nc + n];
    }
}

__global__ void gate_mul_kernel(const float* __restrict__ o,
                                const float* __restrict__ g,
                                __bf16* __restrict__ out, long long n)
{
    long long i = (long long)blockIdx.x * 256 + threadIdx.x;
    if (i < n) out[i] = (__bf16)(o[i] * sigmf(g[i]));
}

__global__ void conv_bf16_kernel(const float* __restrict__ in,
                                 __bf16* __restrict__ out, long long n)
{
    long long i = (long long)blockIdx.x * 256 + threadIdx.x;
    if (i < n) out[i] = (__bf16)in[i];
}

// ---------------------------------------------------------------------------
extern "C" void kernel_launch(void* const* d_in, const int* in_sizes, int n_in,
                              void* d_out, int out_size, void* d_ws,
                              size_t ws_size, hipStream_t stream)
{
    (void)in_sizes; (void)n_in; (void)out_size; (void)ws_size;
    const int NN = 65536;          // N*N, N = 256
    const float* s_in = (const float*)d_in[0];
    const float* z_in = (const float*)d_in[1];
    auto W = [&](int i) { return (const float*)d_in[i]; };

    char* base = (char*)d_ws;
    size_t off = 0;
    auto alloc = [&](size_t bytes) -> char* {
        off = (off + 255) & ~(size_t)255;
        char* r = base + off; off += bytes; return r;
    };

    // ---- workspace carve-out ----
    __bf16* wtp  = (__bf16*)alloc(8u << 20);          size_t wto = 0;
    float*  zf   = (float*) alloc((size_t)NN * 128 * 4);
    float*  sf   = (float*) alloc((size_t)256 * 384 * 4);
    __bf16* xb   = (__bf16*)alloc((size_t)NN * 128 * 2);
    __bf16* tnb  = (__bf16*)alloc((size_t)NN * 128 * 2);
    __bf16* acm  = (__bf16*)alloc((size_t)NN * 128 * 2);
    __bf16* bcm  = (__bf16*)alloc((size_t)NN * 128 * 2);
    __bf16* ob   = (__bf16*)alloc((size_t)NN * 128 * 2);
    __bf16* big0 = (__bf16*)alloc((size_t)NN * 512 * 2);  // gateb / q / silu-h1
    __bf16* big1 = (__bf16*)alloc((size_t)NN * 512 * 2);  // k,v / gated-h2
    float*  f0   = (float*) alloc((size_t)NN * 128 * 4);  // t_cm / attn-out
    float*  f1   = (float*) alloc((size_t)NN * 128 * 4);  // gate f32
    float*  bsf  = (float*) alloc((size_t)NN * 16 * 4);   // attn bias buf
    __bf16* sb   = (__bf16*)alloc((size_t)256 * 384 * 2);
    __bf16* qsb  = (__bf16*)alloc((size_t)256 * 384 * 2);
    __bf16* ksb  = (__bf16*)alloc((size_t)256 * 384 * 2);
    __bf16* vsb  = (__bf16*)alloc((size_t)256 * 384 * 2);
    float*  gsf  = (float*) alloc((size_t)256 * 384 * 4);
    float*  osf  = (float*) alloc((size_t)256 * 384 * 4);
    __bf16* osb  = (__bf16*)alloc((size_t)256 * 384 * 2);
    __bf16* hs1  = (__bf16*)alloc((size_t)256 * 1536 * 2);
    __bf16* hs2  = (__bf16*)alloc((size_t)256 * 1536 * 2);

    auto wconv = [&](int idx, int K, int Nc) -> __bf16* {
        __bf16* dst = wtp + wto; wto += (size_t)K * Nc;
        int n = K * Nc;
        wt_kernel<<<(n + 255) / 256, 256, 0, stream>>>(W(idx), dst, K, Nc);
        return dst;
    };
    auto gemm = [&](const __bf16* A, const __bf16* Bt, int M, int Nc, int K,
                    int lda, int batch, long long sA, long long sB,
                    long long sC, float* c0, __bf16* c1, __bf16* c2,
                    const float* e0, const __bf16* e1, const float* bias,
                    int ldc, int lde, int ep, int tf) {
        int tiles = (M >> 5) * ((Nc + 31) >> 5);
        dim3 g((tiles + 7) / 8, batch);
        gemm_bf16_wmma<<<g, 256, 0, stream>>>(A, Bt, M, Nc, K, lda, sA, sB, sC,
                                              c0, c1, c2, e0, e1, bias,
                                              ldc, lde, ep, tf);
    };
    auto ln = [&](const float* in, const float* w, const float* b,
                  __bf16* out, int M, int C, int mode) {
        ln_kernel<<<(M + 7) / 8, 256, 0, stream>>>(in, w, b, out, M, C, mode);
    };

    // ---- weights -> bf16 [Nc][K] ----
    __bf16 *tmo_pin = wconv(4, 128, 256),  *tmo_gin = wconv(5, 128, 256);
    __bf16 *tmo_pout = wconv(8, 128, 128), *tmo_gout = wconv(9, 128, 128);
    __bf16 *tmi_pin = wconv(12, 128, 256), *tmi_gin = wconv(13, 128, 256);
    __bf16 *tmi_pout = wconv(16, 128, 128), *tmi_gout = wconv(17, 128, 128);
    __bf16 *tas_bw = wconv(20, 128, 4),  *tas_qw = wconv(21, 128, 128);
    __bf16 *tas_kw = wconv(22, 128, 128), *tas_vw = wconv(23, 128, 128);
    __bf16 *tas_gw = wconv(24, 128, 128), *tas_ow = wconv(25, 128, 128);
    __bf16 *tae_bw = wconv(28, 128, 4),  *tae_qw = wconv(29, 128, 128);
    __bf16 *tae_kw = wconv(30, 128, 128), *tae_vw = wconv(31, 128, 128);
    __bf16 *tae_gw = wconv(32, 128, 128), *tae_ow = wconv(33, 128, 128);
    __bf16 *tz_f1 = wconv(36, 128, 512), *tz_f2 = wconv(37, 128, 512);
    __bf16 *tz_f3 = wconv(38, 512, 128);
    __bf16 *apb_qw = wconv(39, 384, 384), *apb_kw = wconv(41, 384, 384);
    __bf16 *apb_vw = wconv(42, 384, 384), *apb_gw = wconv(43, 384, 384);
    __bf16 *apb_zw = wconv(46, 128, 16),  *apb_ow = wconv(47, 384, 384);
    __bf16 *ts_f1 = wconv(50, 384, 1536), *ts_f2 = wconv(51, 384, 1536);
    __bf16 *ts_f3 = wconv(52, 1536, 384);

    // ---- state init ----
    hipMemcpyAsync(zf, z_in, (size_t)NN * 128 * 4, hipMemcpyDeviceToDevice, stream);
    hipMemcpyAsync(sf, s_in, (size_t)256 * 384 * 4, hipMemcpyDeviceToDevice, stream);

    // ---- triangle multiplications (outgoing then incoming) ----
    auto tri_mul = [&](int pi, __bf16* pin, __bf16* gin, __bf16* pout,
                       __bf16* gout, int T) {
        ln(zf, W(pi + 0), W(pi + 1), xb, NN, 128, 0);
        gemm(xb, gin, NN, 256, 128, 128, 1, 0, 0, 0,
             nullptr, big0, nullptr, nullptr, nullptr, nullptr,
             256, 0, EP_SIG_BF16, 0);
        gemm(xb, pin, NN, 256, 128, 128, 1, 0, 0, 0,
             nullptr, acm, bcm, nullptr, big0, nullptr,
             0, 256, EP_GATE_CM, T);
        // 128 batched per-channel GEMMs: t_c = A_c * B_c^T
        gemm(acm, bcm, 256, 256, 256, 256, 128, 65536, 65536, 65536,
             f0, nullptr, nullptr, nullptr, nullptr, nullptr,
             256, 0, EP_F32, 0);
        ln(f0, W(pi + 4), W(pi + 5), tnb, NN, 128, 2);
        gemm(xb, gout, NN, 128, 128, 128, 1, 0, 0, 0,
             f1, nullptr, nullptr, nullptr, nullptr, nullptr,
             128, 0, EP_F32, 0);
        gemm(tnb, pout, NN, 128, 128, 128, 1, 0, 0, 0,
             zf, nullptr, nullptr, f1, nullptr, nullptr,
             128, 128, EP_GATE_RESID, 0);
    };
    tri_mul(2,  tmo_pin, tmo_gin, tmo_pout, tmo_gout, 0);   // outgoing
    tri_mul(10, tmi_pin, tmi_gin, tmi_pout, tmi_gout, 1);   // incoming

    // ---- triangle attentions (starting then ending) ----
    __bf16* qtb = big0;
    __bf16* ktb = big1;
    __bf16* vtb = big1 + (size_t)NN * 128;
    auto tri_attn = [&](int pi, __bf16* bw, __bf16* qw, __bf16* kw,
                        __bf16* vw, __bf16* gw, __bf16* ow, int ending) {
        ln(zf, W(pi + 0), W(pi + 1), xb, NN, 128, ending ? 1 : 0);
        gemm(xb, bw, NN, 4, 128, 128, 1, 0, 0, 0,
             bsf, nullptr, nullptr, nullptr, nullptr, nullptr,
             4, 0, EP_F32, 0);
        gemm(xb, qw, NN, 128, 128, 128, 1, 0, 0, 0,
             nullptr, qtb, nullptr, nullptr, nullptr, nullptr,
             128, 0, EP_BF16, 0);
        gemm(xb, kw, NN, 128, 128, 128, 1, 0, 0, 0,
             nullptr, ktb, nullptr, nullptr, nullptr, nullptr,
             128, 0, EP_BF16, 0);
        gemm(xb, vw, NN, 128, 128, 128, 1, 0, 0, 0,
             nullptr, vtb, nullptr, nullptr, nullptr, nullptr,
             128, 0, EP_BF16, 0);
        gemm(xb, gw, NN, 128, 128, 128, 1, 0, 0, 0,
             f1, nullptr, nullptr, nullptr, nullptr, nullptr,
             128, 0, EP_F32, 0);
        attn_kernel<<<256 * 4, 256, 0, stream>>>(
            qtb, ktb, vtb, 128, (long long)256 * 128, 32, 4, 32,
            0.17677669529663687f, bsf, 4, f0, (long long)256 * 128, 32, 128);
        long long n = (long long)NN * 128;
        gate_mul_kernel<<<(unsigned)((n + 255) / 256), 256, 0, stream>>>(
            f0, f1, ob, n);
        gemm(ob, ow, NN, 128, 128, 128, 1, 0, 0, 0,
             zf, nullptr, nullptr, nullptr, nullptr, nullptr,
             128, 0, ending ? EP_RESID_T : EP_RESID, 0);
    };
    tri_attn(18, tas_bw, tas_qw, tas_kw, tas_vw, tas_gw, tas_ow, 0);
    tri_attn(26, tae_bw, tae_qw, tae_kw, tae_vw, tae_gw, tae_ow, 1);

    // ---- pair transition ----
    ln(zf, W(34), W(35), xb, NN, 128, 0);
    gemm(xb, tz_f1, NN, 512, 128, 128, 1, 0, 0, 0,
         nullptr, big0, nullptr, nullptr, nullptr, nullptr,
         512, 0, EP_SILU_BF16, 0);
    gemm(xb, tz_f2, NN, 512, 128, 128, 1, 0, 0, 0,
         nullptr, big1, nullptr, nullptr, big0, nullptr,
         512, 512, EP_MUL_BF16, 0);
    gemm(big1, tz_f3, NN, 128, 512, 512, 1, 0, 0, 0,
         zf, nullptr, nullptr, nullptr, nullptr, nullptr,
         128, 0, EP_RESID, 0);

    // ---- attention with pair bias (updates s) ----
    conv_bf16_kernel<<<(98304 + 255) / 256, 256, 0, stream>>>(sf, sb, 98304);
    gemm(sb, apb_qw, 256, 384, 384, 384, 1, 0, 0, 0,
         nullptr, qsb, nullptr, nullptr, nullptr, W(40),
         384, 0, EP_BF16, 0);
    gemm(sb, apb_kw, 256, 384, 384, 384, 1, 0, 0, 0,
         nullptr, ksb, nullptr, nullptr, nullptr, nullptr,
         384, 0, EP_BF16, 0);
    gemm(sb, apb_vw, 256, 384, 384, 384, 1, 0, 0, 0,
         nullptr, vsb, nullptr, nullptr, nullptr, nullptr,
         384, 0, EP_BF16, 0);
    gemm(sb, apb_gw, 256, 384, 384, 384, 1, 0, 0, 0,
         gsf, nullptr, nullptr, nullptr, nullptr, nullptr,
         384, 0, EP_F32, 0);
    ln(zf, W(44), W(45), xb, NN, 128, 0);
    gemm(xb, apb_zw, NN, 16, 128, 128, 1, 0, 0, 0,
         bsf, nullptr, nullptr, nullptr, nullptr, nullptr,
         16, 0, EP_F32, 0);
    attn_kernel<<<16, 256, 0, stream>>>(
        qsb, ksb, vsb, 384, 0, 24, 16, 24, 0.2041241452319315f,
        bsf, 16, osf, 0, 24, 384);
    gate_mul_kernel<<<(98304 + 255) / 256, 256, 0, stream>>>(
        osf, gsf, osb, 98304);
    gemm(osb, apb_ow, 256, 384, 384, 384, 1, 0, 0, 0,
         sf, nullptr, nullptr, nullptr, nullptr, nullptr,
         384, 0, EP_RESID, 0);

    // ---- single transition ----
    ln(sf, W(48), W(49), sb, 256, 384, 0);
    gemm(sb, ts_f1, 256, 1536, 384, 384, 1, 0, 0, 0,
         nullptr, hs1, nullptr, nullptr, nullptr, nullptr,
         1536, 0, EP_SILU_BF16, 0);
    gemm(sb, ts_f2, 256, 1536, 384, 384, 1, 0, 0, 0,
         nullptr, hs2, nullptr, nullptr, hs1, nullptr,
         1536, 1536, EP_MUL_BF16, 0);
    gemm(hs2, ts_f3, 256, 384, 1536, 1536, 1, 0, 0, 0,
         sf, nullptr, nullptr, nullptr, nullptr, nullptr,
         384, 0, EP_RESID, 0);

    // ---- outputs: (s, z) flat ----
    hipMemcpyAsync(d_out, sf, (size_t)256 * 384 * 4,
                   hipMemcpyDeviceToDevice, stream);
    hipMemcpyAsync((float*)d_out + 98304, zf, (size_t)NN * 128 * 4,
                   hipMemcpyDeviceToDevice, stream);
}
